// DeepseekV2Attention_47802986004844
// MI455X (gfx1250) — matrix operand order repaired and checked by reading, not compiled, not run on hardware
//
#include <hip/hip_runtime.h>
#include <hip/hip_bf16.h>

// ---------------- problem constants ----------------
#define T_SEQ   4096
#define HIDDEN  2048
#define NH      16
#define QK_NOPE 128
#define QK_ROPE 64
#define VDIM    128
#define QLORA   1536
#define KVLORA  512
#define QKHEAD  192   // 128 + 64

// ---------------- types ----------------
typedef __attribute__((ext_vector_type(16))) __bf16 v16bf;
typedef __attribute__((ext_vector_type(8)))  float  v8f;
typedef __attribute__((ext_vector_type(4)))  unsigned int uint32x4;
typedef __attribute__((ext_vector_type(8)))  int int32x8;
typedef __attribute__((ext_vector_type(4)))  int int32x4;

struct bf16x16 { uint4 lo; uint4 hi; };   // 32 bytes == v16bf

__device__ __forceinline__ unsigned short f32_to_bf16(float f) {
    unsigned int u = __builtin_bit_cast(unsigned int, f);
    unsigned int r = u + 0x7FFFu + ((u >> 16) & 1u);   // RNE
    return (unsigned short)(r >> 16);
}

// LDS byte offset of a __shared__ object: low 32 bits of the flat address
// (LDS aperture lives entirely in addr[63:32]; addr[31:0] == LDS offset).
__device__ __forceinline__ unsigned lds_off(const void* p) {
    return (unsigned)(size_t)p;
}

// ---------------- TDM: 2-D bf16 tile load, global -> LDS ----------------
// tile_d0 elems per row (contiguous), tile_d1 rows, global row stride in elems.
// pad_en inserts 8 DWORDs (16 bf16) after every 16 DWORDs (32 bf16) stored,
// i.e. LDS row stride = 48 bf16 when tile_d0 == 32.
__device__ __forceinline__ void tdm_load_tile_bf16(
        unsigned lds_byte_off, const void* gaddr,
        unsigned tile_d0, unsigned tile_d1,
        unsigned stride_elems, bool pad_en) {
    unsigned long long ga = (unsigned long long)(size_t)gaddr;
    uint32x4 g0;
    g0[0] = 1u;                                          // count=1 (user D#)
    g0[1] = lds_byte_off;                                // lds_addr
    g0[2] = (unsigned)(ga & 0xFFFFFFFFu);                // global_addr[31:0]
    g0[3] = (unsigned)((ga >> 32) & 0x01FFFFFFu)         // global_addr[56:32]
          | (2u << 30);                                  // type=2 ("image")
    int32x8 g1;
    unsigned f0 = (1u << 16);                            // data_size=1 (2 bytes)
    if (pad_en) f0 |= (1u << 20)                         // pad_enable
               | (3u << 22)                              // pad_interval: 16 DW
               | (7u << 25);                             // pad_amount:   8 DW
    g1[0] = (int)f0;
    g1[1] = (int)((tile_d0 & 0xFFFFu) << 16);            // tensor_dim0[15:0]
    g1[2] = (int)(((tile_d0 >> 16) & 0xFFFFu)            // tensor_dim0[31:16]
          | ((tile_d1 & 0xFFFFu) << 16));                // tensor_dim1[15:0]
    g1[3] = (int)(((tile_d1 >> 16) & 0xFFFFu)            // tensor_dim1[31:16]
          | ((tile_d0 & 0xFFFFu) << 16));                // tile_dim0
    g1[4] = (int)(tile_d1 & 0xFFFFu);                    // tile_dim1 (tile_dim2=0)
    g1[5] = (int)stride_elems;                           // tensor_dim0_stride lo32
    g1[6] = 0;                                           // stride hi16, dim1_stride lo16
    g1[7] = 0;
    int32x4 z4 = {0, 0, 0, 0};
#if defined(__clang_major__) && (__clang_major__ >= 23)
    int32x8 z8 = {0, 0, 0, 0, 0, 0, 0, 0};
    __builtin_amdgcn_tensor_load_to_lds(g0, g1, z4, z4, z8, 0);
#else
    __builtin_amdgcn_tensor_load_to_lds(g0, g1, z4, z4, 0);
#endif
}

// ---------------- f32 -> bf16 convert (vectorized) ----------------
__global__ __launch_bounds__(256) void cvt_f32_bf16(const float* __restrict__ in,
                                                    unsigned short* __restrict__ out,
                                                    int n4) {
    for (int i = blockIdx.x * blockDim.x + threadIdx.x; i < n4;
         i += gridDim.x * blockDim.x) {
        float4 v = ((const float4*)in)[i];
        ushort4 o;
        o.x = f32_to_bf16(v.x); o.y = f32_to_bf16(v.y);
        o.z = f32_to_bf16(v.z); o.w = f32_to_bf16(v.w);
        ((ushort4*)out)[i] = o;
    }
}

// ---------------- generic bf16 WMMA GEMM: C(f32) = A(bf16,MxK) * B(bf16,KxN) ----------------
// block tile 128x64, 8 waves (4x2), each wave 32x32 (2x2 WMMA accums), K step 32.
// Double-buffered pipeline: TDM streams A tile i+1 while WMMAs consume tile i.
#define G_BM 128
#define G_BN 64
#define G_BK 32
#define LDA_S 48   // bf16 elems per LDS row of A (32 + 16 pad from TDM)
#define LDB_S 48   // bf16 elems per LDS row of Bt

__global__ __launch_bounds__(256) void gemm_bf16f32(
        const unsigned short* __restrict__ A,
        const unsigned short* __restrict__ B,
        float* __restrict__ C, int M, int N, int K) {
    __shared__ unsigned short sA[2][G_BM * LDA_S];
    __shared__ unsigned short sBt[2][G_BN * LDB_S];

    const int tid  = threadIdx.x;
    const int wave = tid >> 5;
    const int lane = tid & 31;
    const int half = lane >> 4;
    const int l16  = lane & 15;
    const int wy   = wave >> 1;   // 0..3
    const int wx   = wave & 1;    // 0..1
    const int bm   = blockIdx.y * G_BM;
    const int bn   = blockIdx.x * G_BN;

    auto stageB = [&](int buf, int k0) {
        for (int c = tid; c < (G_BK * G_BN / 8); c += 256) {
            int r  = c >> 3;          // k row 0..31
            int n8 = (c & 7) * 8;     // n col
            const unsigned short* gp = B + (size_t)(k0 + r) * N + bn + n8;
            uint4 d = *(const uint4*)gp;
            if (k0 + G_BK < K)
                __builtin_prefetch(gp + (size_t)G_BK * N, 0, 3);
            const unsigned short* ds = (const unsigned short*)&d;
            #pragma unroll
            for (int j = 0; j < 8; ++j)
                sBt[buf][(n8 + j) * LDB_S + r] = ds[j];
        }
    };

    const v8f vzero = {0.f,0.f,0.f,0.f,0.f,0.f,0.f,0.f};
    v8f acc[2][2];
    acc[0][0] = vzero; acc[0][1] = vzero; acc[1][0] = vzero; acc[1][1] = vzero;

    const int nk = K / G_BK;
    // prologue: tile 0 in flight
    if (wave == 0)
        tdm_load_tile_bf16(lds_off(&sA[0][0]), A + (size_t)bm * K,
                           G_BK, G_BM, (unsigned)K, true);
    stageB(0, 0);

    for (int i = 0; i < nk; ++i) {
        const int cur = i & 1;
        if (wave == 0)
            __builtin_amdgcn_s_wait_tensorcnt(0);   // tile i resident
        __syncthreads();                             // publish A(i) + B(i); retire reads of buf i-1
        if (i + 1 < nk) {
            const int nxt = cur ^ 1;
            const int k0n = (i + 1) * G_BK;
            if (wave == 0)
                tdm_load_tile_bf16(lds_off(&sA[nxt][0]), A + (size_t)bm * K + k0n,
                                   G_BK, G_BM, (unsigned)K, true);
            stageB(nxt, k0n);                        // overlaps with compute below
        }

        #pragma unroll
        for (int mi = 0; mi < 2; ++mi) {
            // A fragment: lane row, K chunks {half*8 .. +7} and {half*8+16 .. +23}
            int row = wy * 32 + mi * 16 + l16;
            bf16x16 af;
            af.lo = *(const uint4*)(&sA[cur][row * LDA_S + half * 8]);
            af.hi = *(const uint4*)(&sA[cur][row * LDA_S + half * 8 + 16]);
            v16bf a = __builtin_bit_cast(v16bf, af);
            #pragma unroll
            for (int ni = 0; ni < 2; ++ni) {
                // B fragment: lane col, K chunks {half*8 .. +7} and {half*8+16 .. +23}
                int col = wx * 32 + ni * 16 + l16;
                bf16x16 bf;
                bf.lo = *(const uint4*)(&sBt[cur][col * LDB_S + half * 8]);
                bf.hi = *(const uint4*)(&sBt[cur][col * LDB_S + half * 8 + 16]);
                v16bf b = __builtin_bit_cast(v16bf, bf);
                acc[mi][ni] = __builtin_amdgcn_wmma_f32_16x16x32_bf16(
                    false, a, false, b, (short)0, acc[mi][ni], false, false);
            }
        }
    }

    // C layout: VGPR r -> row r + 8*half ; lane -> col l16
    #pragma unroll
    for (int mi = 0; mi < 2; ++mi)
        #pragma unroll
        for (int ni = 0; ni < 2; ++ni) {
            int col = bn + wx * 32 + ni * 16 + l16;
            #pragma unroll
            for (int r = 0; r < 8; ++r) {
                int row = bm + wy * 32 + mi * 16 + half * 8 + r;
                C[(size_t)row * N + col] = acc[mi][ni][r];
            }
        }
}

// ---------------- RMSNorm: f32 in (row stride in_stride) -> bf16 out (row stride C) ----------------
__global__ __launch_bounds__(256) void rmsnorm_bf16(const float* __restrict__ in,
                                                    const float* __restrict__ w,
                                                    unsigned short* __restrict__ out,
                                                    int C, int in_stride) {
    const int t = blockIdx.x;
    const float* x = in + (size_t)t * in_stride;
    float s = 0.f;
    for (int i = threadIdx.x; i < C; i += 256) { float v = x[i]; s += v * v; }
    s += __shfl_xor(s, 16, 32);
    s += __shfl_xor(s, 8, 32);
    s += __shfl_xor(s, 4, 32);
    s += __shfl_xor(s, 2, 32);
    s += __shfl_xor(s, 1, 32);
    __shared__ float red[8];
    if ((threadIdx.x & 31) == 0) red[threadIdx.x >> 5] = s;
    __syncthreads();
    if (threadIdx.x == 0) {
        float tot = 0.f;
        #pragma unroll
        for (int i = 0; i < 8; ++i) tot += red[i];
        red[0] = tot;
    }
    __syncthreads();
    const float rms = rsqrtf(red[0] / (float)C + 1e-6f);
    for (int i = threadIdx.x; i < C; i += 256)
        out[(size_t)t * C + i] = f32_to_bf16(x[i] * rms * w[i]);
}

// ---------------- Q prep: copy nope + RoPE(pe) -> bf16 [T][NH][192] ----------------
__global__ __launch_bounds__(256) void q_prep(const float* __restrict__ q,
                                              const int* __restrict__ pos,
                                              unsigned short* __restrict__ out) {
    const int t = blockIdx.x;
    const float p = (float)pos[t];
    const float* qt = q + (size_t)t * (NH * QKHEAD);
    unsigned short* ot = out + (size_t)t * (NH * QKHEAD);
    for (int i = threadIdx.x; i < NH * QK_NOPE; i += 256) {
        int h = i >> 7, d = i & 127;
        ot[h * QKHEAD + d] = f32_to_bf16(qt[h * QKHEAD + d]);
    }
    for (int i = threadIdx.x; i < NH * (QK_ROPE / 2); i += 256) {
        int h = i >> 5, j = i & 31;
        float inv = __powf(10000.0f, -(float)j / 32.0f);  // theta^(-2j/64)
        float ang = p * inv;
        float cv = __cosf(ang), sv = __sinf(ang);
        float x1 = qt[h * QKHEAD + QK_NOPE + 2 * j];
        float x2 = qt[h * QKHEAD + QK_NOPE + 2 * j + 1];
        ot[h * QKHEAD + QK_NOPE + 2 * j]     = f32_to_bf16(x1 * cv - x2 * sv);
        ot[h * QKHEAD + QK_NOPE + 2 * j + 1] = f32_to_bf16(x2 * cv + x1 * sv);
    }
}

// ---------------- KV prep: split kv, rope+broadcast k_pe -> K bf16 [T][NH][192], V bf16 [T][NH][128] ----------------
__global__ __launch_bounds__(256) void kv_prep(const float* __restrict__ kv,
                                               const float* __restrict__ kva,
                                               const int* __restrict__ pos,
                                               unsigned short* __restrict__ kout,
                                               unsigned short* __restrict__ vout) {
    const int t = blockIdx.x;
    const float p = (float)pos[t];
    const float* kvt = kv + (size_t)t * (NH * (QK_NOPE + VDIM));
    unsigned short* kt = kout + (size_t)t * (NH * QKHEAD);
    unsigned short* vt = vout + (size_t)t * (NH * VDIM);
    __shared__ unsigned short kpe[QK_ROPE];
    if (threadIdx.x < QK_ROPE / 2) {
        int j = threadIdx.x;
        float inv = __powf(10000.0f, -(float)j / 32.0f);
        float ang = p * inv;
        float cv = __cosf(ang), sv = __sinf(ang);
        float x1 = kva[(size_t)t * (KVLORA + QK_ROPE) + KVLORA + 2 * j];
        float x2 = kva[(size_t)t * (KVLORA + QK_ROPE) + KVLORA + 2 * j + 1];
        kpe[2 * j]     = f32_to_bf16(x1 * cv - x2 * sv);
        kpe[2 * j + 1] = f32_to_bf16(x2 * cv + x1 * sv);
    }
    __syncthreads();
    for (int i = threadIdx.x; i < NH * 128; i += 256) {
        int h = i >> 7, d = i & 127;
        kt[h * QKHEAD + d] = f32_to_bf16(kvt[h * 256 + d]);
        vt[h * VDIM + d]   = f32_to_bf16(kvt[h * 256 + 128 + d]);
    }
    for (int i = threadIdx.x; i < NH * QK_ROPE; i += 256) {
        int h = i >> 6, d = i & 63;
        kt[h * QKHEAD + QK_NOPE + d] = kpe[d];
    }
}

// ---------------- Flash attention: 1 WG = 128 q rows x 1 head; 8 waves x 16 rows ----------------
// K tile: 6 TDM chunk loads of 64x32 bf16, each chunk stored [64][48] (TDM pad).
// Double-buffered: TDM streams K(i+1) and threads stage V(i+1) while tile i computes.
#define FK_CH 6
#define FK_CS 48              // elems per row within a chunk (32 + 16 pad)
#define FK_CHUNK (64 * FK_CS) // elems per chunk region
#define FV_S 72    // LDS row stride for Vt tile (64 + pad, 144B, 16B aligned)
#define FP_S 72    // LDS row stride for P staging

__global__ __launch_bounds__(256) void flash_attn(
        const unsigned short* __restrict__ Q,
        const unsigned short* __restrict__ Kb,
        const unsigned short* __restrict__ Vb,
        unsigned short* __restrict__ O) {
    __shared__ unsigned short sK[2][FK_CH * FK_CHUNK];
    __shared__ unsigned short sVt[2][128 * FV_S];
    __shared__ unsigned short sP[8 * 16 * FP_S];

    const int tid  = threadIdx.x;
    const int wave = tid >> 5;
    const int lane = tid & 31;
    const int half = lane >> 4;
    const int l16  = lane & 15;
    const int h    = blockIdx.y;
    const int qb   = blockIdx.x * 128;
    const int qrow = qb + wave * 16 + l16;

    auto issueK = [&](int buf, int kbase) {
        #pragma unroll
        for (int c = 0; c < FK_CH; ++c)
            tdm_load_tile_bf16(lds_off(&sK[buf][0]) + c * FK_CHUNK * 2,
                               Kb + (((size_t)kbase * NH + h) * QKHEAD) + c * 32,
                               32, 64, NH * QKHEAD, true);
    };
    auto stageV = [&](int buf, int kbase) {
        for (int c = tid; c < 64 * 128 / 8; c += 256) {
            int r  = c / 16;
            int d8 = (c % 16) * 8;
            uint4 d = *(const uint4*)(Vb + (((size_t)(kbase + r)) * NH + h) * VDIM + d8);
            const unsigned short* ds = (const unsigned short*)&d;
            #pragma unroll
            for (int j = 0; j < 8; ++j)
                sVt[buf][(d8 + j) * FV_S + r] = ds[j];
        }
    };

    // Q fragments for this wave's 16 rows: 6 chunks of K=32
    v16bf qf[6];
    const unsigned short* qp = Q + ((size_t)qrow * NH + h) * QKHEAD;
    #pragma unroll
    for (int c = 0; c < 6; ++c) {
        bf16x16 f;
        f.lo = *(const uint4*)(qp + c * 32 + half * 8);
        f.hi = *(const uint4*)(qp + c * 32 + half * 8 + 16);
        qf[c] = __builtin_bit_cast(v16bf, f);
    }

    const v8f vzero = {0.f,0.f,0.f,0.f,0.f,0.f,0.f,0.f};
    v8f oacc[8];
    #pragma unroll
    for (int dt = 0; dt < 8; ++dt) oacc[dt] = vzero;
    float mrow[8], lrow[8];
    #pragma unroll
    for (int r = 0; r < 8; ++r) { mrow[r] = -1e30f; lrow[r] = 0.f; }

    const float scale = 0.07216878364870322f;   // 1/sqrt(192)
    const int nkb = 2 * blockIdx.x + 2;         // causal: keys up to qb+127

    // prologue: tile 0 in flight
    if (wave == 0) issueK(0, 0);
    stageV(0, 0);

    for (int kb = 0; kb < nkb; ++kb) {
        const int cur = kb & 1;
        const int kbase = kb * 64;
        if (wave == 0)
            __builtin_amdgcn_s_wait_tensorcnt(0);   // K tile kb resident
        __syncthreads();                             // publish K(kb) + V(kb)
        if (kb + 1 < nkb) {
            const int nxt = cur ^ 1;
            if (wave == 0) issueK(nxt, kbase + 64);
            stageV(nxt, kbase + 64);                 // overlaps with compute below
        }

        // S = Q * K^T : 4 column tiles of 16 keys
        v8f sacc[4];
        #pragma unroll
        for (int jt = 0; jt < 4; ++jt) sacc[jt] = vzero;
        #pragma unroll
        for (int jt = 0; jt < 4; ++jt) {
            const int j = jt * 16 + l16;
            #pragma unroll
            for (int c = 0; c < 6; ++c) {
                const unsigned short* ck = &sK[cur][c * FK_CHUNK];
                bf16x16 f;
                f.lo = *(const uint4*)(&ck[j * FK_CS + half * 8]);
                f.hi = *(const uint4*)(&ck[j * FK_CS + half * 8 + 16]);
                v16bf b = __builtin_bit_cast(v16bf, f);
                sacc[jt] = __builtin_amdgcn_wmma_f32_16x16x32_bf16(
                    false, qf[c], false, b, (short)0, sacc[jt], false, false);
            }
        }

        // scale + causal mask + running row max
        float newmax[8];
        #pragma unroll
        for (int r = 0; r < 8; ++r) newmax[r] = mrow[r];
        #pragma unroll
        for (int jt = 0; jt < 4; ++jt) {
            const int col = kbase + jt * 16 + l16;
            #pragma unroll
            for (int r = 0; r < 8; ++r) {
                const int row = qb + wave * 16 + half * 8 + r;
                float v = sacc[jt][r] * scale;
                v = (col <= row) ? v : -1e30f;
                sacc[jt][r] = v;
                newmax[r] = fmaxf(newmax[r], v);
            }
        }
        #pragma unroll
        for (int r = 0; r < 8; ++r) {
            float v = newmax[r];
            v = fmaxf(v, __shfl_xor(v, 1, 32));
            v = fmaxf(v, __shfl_xor(v, 2, 32));
            v = fmaxf(v, __shfl_xor(v, 4, 32));
            v = fmaxf(v, __shfl_xor(v, 8, 32));
            newmax[r] = v;
        }

        // exp, partial sums, stage P (bf16) into per-wave LDS for transpose
        unsigned short* pw = &sP[wave * 16 * FP_S];
        float psum[8];
        #pragma unroll
        for (int r = 0; r < 8; ++r) psum[r] = 0.f;
        #pragma unroll
        for (int jt = 0; jt < 4; ++jt) {
            #pragma unroll
            for (int r = 0; r < 8; ++r) {
                float e = __expf(sacc[jt][r] - newmax[r]);
                psum[r] += e;
                pw[(half * 8 + r) * FP_S + jt * 16 + l16] = f32_to_bf16(e);
            }
        }
        #pragma unroll
        for (int r = 0; r < 8; ++r) {
            float ps = psum[r];
            ps += __shfl_xor(ps, 1, 32);
            ps += __shfl_xor(ps, 2, 32);
            ps += __shfl_xor(ps, 4, 32);
            ps += __shfl_xor(ps, 8, 32);
            float corr = __expf(mrow[r] - newmax[r]);
            lrow[r] = lrow[r] * corr + ps;
            mrow[r] = newmax[r];
            #pragma unroll
            for (int dt = 0; dt < 8; ++dt) oacc[dt][r] *= corr;
        }

        // O += P * V : P is 16x64 (2 K-chunks of 32), V tiles along d
        #pragma unroll
        for (int c = 0; c < 2; ++c) {
            bf16x16 pf;
            pf.lo = *(const uint4*)(&pw[l16 * FP_S + c * 32 + half * 8]);
            pf.hi = *(const uint4*)(&pw[l16 * FP_S + c * 32 + half * 8 + 16]);
            v16bf pa = __builtin_bit_cast(v16bf, pf);
            #pragma unroll
            for (int dt = 0; dt < 8; ++dt) {
                const int d = dt * 16 + l16;
                bf16x16 vf;
                vf.lo = *(const uint4*)(&sVt[cur][d * FV_S + c * 32 + half * 8]);
                vf.hi = *(const uint4*)(&sVt[cur][d * FV_S + c * 32 + half * 8 + 16]);
                v16bf vb = __builtin_bit_cast(v16bf, vf);
                oacc[dt] = __builtin_amdgcn_wmma_f32_16x16x32_bf16(
                    false, pa, false, vb, (short)0, oacc[dt], false, false);
            }
        }
    }

    // normalize and write bf16 attn output [T][NH*VDIM]
    #pragma unroll
    for (int dt = 0; dt < 8; ++dt) {
        #pragma unroll
        for (int r = 0; r < 8; ++r) {
            const int row = qb + wave * 16 + half * 8 + r;
            float v = oacc[dt][r] / lrow[r];
            O[(size_t)row * (NH * VDIM) + h * VDIM + dt * 16 + l16] = f32_to_bf16(v);
        }
    }
}

// ---------------- host orchestration ----------------
extern "C" void kernel_launch(void* const* d_in, const int* in_sizes, int n_in,
                              void* d_out, int out_size, void* d_ws, size_t ws_size,
                              hipStream_t stream) {
    const float* hidden = (const float*)d_in[0];
    const int*   pos    = (const int*)d_in[1];
    const float* Wqa    = (const float*)d_in[2];
    const float* q_ln   = (const float*)d_in[3];
    const float* Wqb    = (const float*)d_in[4];
    const float* Wkva   = (const float*)d_in[5];
    const float* kv_ln  = (const float*)d_in[6];
    const float* Wkvb   = (const float*)d_in[7];
    const float* Wo     = (const float*)d_in[8];
    float* out = (float*)d_out;
    char* ws = (char*)d_ws;

    size_t off = 0;
    auto alloc = [&](size_t bytes) { size_t o = off; off += (bytes + 255) & ~(size_t)255; return o; };

    const size_t hid_bf  = alloc((size_t)T_SEQ * HIDDEN * 2);
    const size_t wqa_bf  = alloc((size_t)HIDDEN * QLORA * 2);
    const size_t wqb_bf  = alloc((size_t)QLORA * NH * QKHEAD * 2);
    const size_t wkva_bf = alloc((size_t)HIDDEN * (KVLORA + QK_ROPE) * 2);
    const size_t wkvb_bf = alloc((size_t)KVLORA * NH * 256 * 2);
    const size_t wo_bf   = alloc((size_t)NH * VDIM * HIDDEN * 2);
    const size_t qa_f    = alloc((size_t)T_SEQ * QLORA * 4);
    const size_t qlat_bf = alloc((size_t)T_SEQ * QLORA * 2);
    const size_t q_f     = alloc((size_t)T_SEQ * NH * QKHEAD * 4);
    const size_t q_bf    = alloc((size_t)T_SEQ * NH * QKHEAD * 2);
    const size_t kva_f   = alloc((size_t)T_SEQ * (KVLORA + QK_ROPE) * 4);
    const size_t kvlat_bf= alloc((size_t)T_SEQ * KVLORA * 2);
    const size_t kv_f    = alloc((size_t)T_SEQ * NH * 256 * 4);
    const size_t k_bf    = alloc((size_t)T_SEQ * NH * QKHEAD * 2);
    const size_t v_bf    = alloc((size_t)T_SEQ * NH * VDIM * 2);
    const size_t attn_bf = alloc((size_t)T_SEQ * NH * VDIM * 2);
    (void)ws_size; (void)n_in; (void)in_sizes; (void)out_size;

    auto U16 = [&](size_t o) { return (unsigned short*)(ws + o); };
    auto F32 = [&](size_t o) { return (float*)(ws + o); };

    // 1) bf16 conversions
    auto cvt = [&](const float* src, size_t dsto, size_t n) {
        int n4 = (int)(n / 4);
        int blocks = (n4 + 255) / 256; if (blocks > 4096) blocks = 4096;
        cvt_f32_bf16<<<blocks, 256, 0, stream>>>(src, U16(dsto), n4);
    };
    cvt(hidden, hid_bf,  (size_t)T_SEQ * HIDDEN);
    cvt(Wqa,    wqa_bf,  (size_t)HIDDEN * QLORA);
    cvt(Wqb,    wqb_bf,  (size_t)QLORA * NH * QKHEAD);
    cvt(Wkva,   wkva_bf, (size_t)HIDDEN * (KVLORA + QK_ROPE));
    cvt(Wkvb,   wkvb_bf, (size_t)KVLORA * NH * 256);
    cvt(Wo,     wo_bf,   (size_t)NH * VDIM * HIDDEN);

    auto gemm = [&](size_t a, size_t b, float* c, int M, int N, int K) {
        dim3 grid(N / G_BN, M / G_BM);
        gemm_bf16f32<<<grid, 256, 0, stream>>>(U16(a), U16(b), c, M, N, K);
    };

    // 2) LoRA-down projections
    gemm(hid_bf, wqa_bf,  F32(qa_f),  T_SEQ, QLORA, HIDDEN);
    gemm(hid_bf, wkva_bf, F32(kva_f), T_SEQ, KVLORA + QK_ROPE, HIDDEN);

    // 3) RMSNorms
    rmsnorm_bf16<<<T_SEQ, 256, 0, stream>>>(F32(qa_f),  q_ln,  U16(qlat_bf),  QLORA,  QLORA);
    rmsnorm_bf16<<<T_SEQ, 256, 0, stream>>>(F32(kva_f), kv_ln, U16(kvlat_bf), KVLORA, KVLORA + QK_ROPE);

    // 4) LoRA-up projections
    gemm(qlat_bf,  wqb_bf,  F32(q_f),  T_SEQ, NH * QKHEAD, QLORA);
    gemm(kvlat_bf, wkvb_bf, F32(kv_f), T_SEQ, NH * 256,    KVLORA);

    // 5) RoPE + bf16 packing of Q, K, V
    q_prep<<<T_SEQ, 256, 0, stream>>>(F32(q_f), pos, U16(q_bf));
    kv_prep<<<T_SEQ, 256, 0, stream>>>(F32(kv_f), F32(kva_f), pos, U16(k_bf), U16(v_bf));

    // 6) flash attention
    dim3 fgrid(T_SEQ / 128, NH);
    flash_attn<<<fgrid, 256, 0, stream>>>(U16(q_bf), U16(k_bf), U16(v_bf), U16(attn_bf));

    // 7) output projection -> d_out (f32)
    gemm(attn_bf, wo_bf, out, T_SEQ, HIDDEN, NH * VDIM);
}
